// TransImg_75565654605953
// MI455X (gfx1250) — matrix operand order repaired and checked by reading, not compiled, not run on hardware
//
#include <hip/hip_runtime.h>
#include <hip/hip_bf16.h>

typedef __attribute__((ext_vector_type(16))) __bf16 v16bf;
typedef __attribute__((ext_vector_type(8)))  float  v8f;

#define NNODE 20000
#define NEDGE 320000

__device__ inline unsigned short f2bf_bits(float f) {
    union { float f; unsigned u; } x; x.f = f;
    unsigned u = x.u;
    unsigned r = (u + 0x7FFFu + ((u >> 16) & 1u)) >> 16;  // round-to-nearest-even
    return (unsigned short)r;
}

// ---------------------------------------------------------------------------
// Tiled GEMM: C[M,Nc] = A[M,K] * W[K,Nc], fp32 in/out, bf16 WMMA inner product.
// 128 threads = 4 waves; block computes a 64x64 tile; wave computes 16x64.
// LDS layouts are arranged so each lane's WMMA fragment is two 16-byte,
// 16B-aligned runs (contiguous for A) -> fragments load as 2x ds_load_b128 (no scalar u16s).
// ---------------------------------------------------------------------------
#define TM 64
#define TN 64
#define TK 32
#define AST 40   // bf16 row stride for As (80 B: 16B-aligned rows)
#define BST 40   // bf16 row stride for Bt (transposed [n][k])

// K permutation for A rows: pos 0..7 = K0..7, pos 8..15 = K16..23,
// pos 16..23 = K8..15, pos 24..31 = K24..31  (matches 16-bit A fragment order)
__device__ inline int aperm(int c) {
    int g = c >> 3;
    return (g == 1) ? c + 8 : (g == 2) ? c - 8 : c;
}

__global__ __launch_bounds__(128) void gemm_bf16_wmma(
    const float* __restrict__ A, const float* __restrict__ W,
    float* __restrict__ C, int M, int K, int Nc) {
    __shared__ __align__(16) unsigned short As[TM * AST];  // [m][perm(k)]
    __shared__ __align__(16) unsigned short Bt[TN * BST];  // [n][k]

    const int tid  = threadIdx.x;
    const int wave = tid >> 5;
    const int lane = tid & 31;
    const int h    = lane >> 4;   // half-wave select
    const int l16  = lane & 15;
    const int m0   = blockIdx.x * TM;
    const int n0   = blockIdx.y * TN;
    const bool inM = (m0 + TM <= M);
    const bool inN = (n0 + TN <= Nc);

    v8f acc[4] = {};

    for (int k0 = 0; k0 < K; k0 += TK) {
        const bool inK = (k0 + TK <= K);

        // ---- stage A 64x32 (f32 -> bf16, K-permuted) ----
        if (inM && inK) {
            for (int i = tid; i < (TM * TK) / 4; i += 128) {   // 4 iters
                int r  = i >> 3;            // 8 float4 per row
                int c4 = (i & 7) * 4;
                float4 v = *(const float4*)&A[(long)(m0 + r) * K + k0 + c4];
                unsigned short* d = &As[r * AST + aperm(c4)];  // group-of-8 aligned
                d[0] = f2bf_bits(v.x); d[1] = f2bf_bits(v.y);
                d[2] = f2bf_bits(v.z); d[3] = f2bf_bits(v.w);
            }
        } else {
            for (int i = tid; i < TM * TK; i += 128) {
                int r = i >> 5, c = i & 31;
                int gr = m0 + r, gc = k0 + c;
                float v = (gr < M && gc < K) ? A[(long)gr * K + gc] : 0.0f;
                As[r * AST + aperm(c)] = f2bf_bits(v);
            }
        }
        // ---- stage B 32x64 transposed -> Bt[n][k] ----
        if (inN && inK) {
            for (int i = tid; i < (TK * TN) / 4; i += 128) {   // 4 iters
                int r  = i >> 4;            // 16 float4 per row
                int c4 = (i & 15) * 4;
                float4 v = *(const float4*)&W[(long)(k0 + r) * Nc + n0 + c4];
                Bt[(c4 + 0) * BST + r] = f2bf_bits(v.x);
                Bt[(c4 + 1) * BST + r] = f2bf_bits(v.y);
                Bt[(c4 + 2) * BST + r] = f2bf_bits(v.z);
                Bt[(c4 + 3) * BST + r] = f2bf_bits(v.w);
            }
        } else {
            for (int i = tid; i < TK * TN; i += 128) {
                int r = i >> 6, c = i & 63;
                int gr = k0 + r, gc = n0 + c;
                float v = (gr < K && gc < Nc) ? W[(long)gr * Nc + gc] : 0.0f;
                Bt[c * BST + r] = f2bf_bits(v);
            }
        }
        __syncthreads();

        union Frag { uint4 q[2]; v16bf v; };

        // A fragment: 32 contiguous bytes at [m row, 16h elems] -> 2x b128
        Frag fa;
        const unsigned short* ar = &As[(wave * 16 + l16) * AST + 16 * h];
        fa.q[0] = *(const uint4*)(ar);
        fa.q[1] = *(const uint4*)(ar + 8);

        #pragma unroll
        for (int j = 0; j < 4; ++j) {
            // B fragment: lane (h, n) needs K = 8h + e (e<8), 16 + 8h + (e-8) (e>=8) in Bt[n][k]
            Frag fb;
            const unsigned short* br = &Bt[(j * 16 + l16) * BST + 8 * h];
            fb.q[0] = *(const uint4*)(br);
            fb.q[1] = *(const uint4*)(br + 16);
            acc[j] = __builtin_amdgcn_wmma_f32_16x16x32_bf16(
                false, fa.v, false, fb.v, (short)0, acc[j], false, false);
        }
        __syncthreads();
    }

    // Store: C/D 16x16 f32 layout: VGPR r -> M = r + 8h, lane -> N = l16
    #pragma unroll
    for (int j = 0; j < 4; ++j) {
        int gc = n0 + j * 16 + l16;
        if (gc >= Nc) continue;
        #pragma unroll
        for (int r = 0; r < 8; ++r) {
            int gr = m0 + wave * 16 + r + 8 * h;
            if (gr < M) C[(long)gr * Nc + gc] = acc[j][r];
        }
    }
}

// ---------------------------------------------------------------------------
// Degree / normalization helpers
// ---------------------------------------------------------------------------
__global__ void k_fill1(float* p, int n) {
    int i = blockIdx.x * blockDim.x + threadIdx.x;
    if (i < n) p[i] = 1.0f;   // self-loop contributes weight 1
}
__global__ void k_deg_accum(const int* __restrict__ dst, const float* __restrict__ attr,
                            float* deg, int E, int use_attr) {
    int e = blockIdx.x * blockDim.x + threadIdx.x;
    if (e < E) atomicAdd(&deg[dst[e]], use_attr ? attr[e] : 1.0f);
}
__global__ void k_deg_to_dinv(float* d, int n) {
    int i = blockIdx.x * blockDim.x + threadIdx.x;
    if (i < n) {
        float v = d[i];
        d[i] = (v > 0.0f) ? rsqrtf(fmaxf(v, 1e-12f)) : 0.0f;
    }
}
__global__ void k_edge_norm(const int* __restrict__ src, const int* __restrict__ dst,
                            const float* __restrict__ attr, const float* __restrict__ dinv,
                            float* norm, int E, int use_attr) {
    int e = blockIdx.x * blockDim.x + threadIdx.x;
    if (e < E) {
        float w = use_attr ? attr[e] : 1.0f;
        norm[e] = dinv[src[e]] * w * dinv[dst[e]];
    }
}

// out[i,f] = dinv[i]^2 * xw[i,f] + b[f]   (self-loop term + bias)
__global__ void k_init_out(const float* __restrict__ xw, const float* __restrict__ dinv,
                           const float* __restrict__ b, float* out, int N, int F) {
    long i = (long)blockIdx.x * blockDim.x + threadIdx.x;
    long total = (long)N * F;
    if (i < total) {
        int row = (int)(i / F), col = (int)(i % F);
        float dv = dinv[row];
        out[i] = dv * dv * xw[i] + b[col];
    }
}

// wave-per-edge scatter: out[dst] += norm * xw[src], float4 vectorized
__global__ __launch_bounds__(256) void k_scatter(
    const float* __restrict__ xw, const int* __restrict__ src,
    const int* __restrict__ dst, const float* __restrict__ norm,
    float* out, int E, int F) {
    int e = blockIdx.x * (blockDim.x >> 5) + (threadIdx.x >> 5);
    int lane = threadIdx.x & 31;
    if (e >= E) return;
    int s = src[e], d = dst[e];
    float w = norm[e];
    const float4* xr = (const float4*)(xw + (long)s * F);
    float* orow = out + (long)d * F;
    int F4 = F >> 2;   // all layer widths are multiples of 4
    for (int i = lane; i < F4; i += 32) {
        if (i + 32 < F4) __builtin_prefetch(&xr[i + 32], 0, 0);  // global_prefetch_b8
        float4 v = xr[i];
        int base = i * 4;
        atomicAdd(&orow[base + 0], w * v.x);
        atomicAdd(&orow[base + 1], w * v.y);
        atomicAdd(&orow[base + 2], w * v.z);
        atomicAdd(&orow[base + 3], w * v.w);
    }
}

__global__ void k_elu(float* p, long n) {
    long i = (long)blockIdx.x * blockDim.x + threadIdx.x;
    if (i < n) {
        float v = p[i];
        p[i] = (v > 0.0f) ? v : expm1f(v);
    }
}

// concat[:, 0:64] = h2 ; concat[:, 64:128] = i2
__global__ void k_concat(const float* __restrict__ h2, const float* __restrict__ i2,
                         float* cat, int N) {
    long i = (long)blockIdx.x * blockDim.x + threadIdx.x;
    long total = (long)N * 128;
    if (i < total) {
        int row = (int)(i >> 7), col = (int)(i & 127);
        cat[i] = (col < 64) ? h2[(long)row * 64 + col] : i2[(long)row * 64 + (col - 64)];
    }
}

// ---------------------------------------------------------------------------
// Host orchestration
// ---------------------------------------------------------------------------
static inline int ceil_div(int a, int b) { return (a + b - 1) / b; }

extern "C" void kernel_launch(void* const* d_in, const int* in_sizes, int n_in,
                              void* d_out, int out_size, void* d_ws, size_t ws_size,
                              hipStream_t stream) {
    const int N = NNODE, E = NEDGE;

    const float* features  = (const float*)d_in[0];
    const float* img_feat  = (const float*)d_in[1];
    const int*   ge        = (const int*)d_in[2];
    const int*   ie        = (const int*)d_in[3];
    const float* gene_attr = (const float*)d_in[4];
    const float* img_attr  = (const float*)d_in[5];
    const int* src_g = ge,      *dst_g = ge + E;
    const int* src_i = ie,      *dst_i = ie + E;

    // params flattened in setup_inputs() insertion order: per layer W then b
    auto Wp = [&](int i) { return (const float*)d_in[6 + 2 * i]; };
    auto Bp = [&](int i) { return (const float*)d_in[7 + 2 * i]; };
    enum { L_conv1, L_conv2, L_conv3, L_conv4, L_img1, L_img2, L_img3, L_img4,
           L_neck, L_neck2, L_c3, L_c4 };
    static const int FI[12] = {3000,512,64,512, 1024,512,64,512, 128,64,64,512};
    static const int FO[12] = {512,64,512,3000, 512,64,512,3000, 64,64,512,3000};

    // workspace layout (f32)
    float* ws = (float*)d_ws;
    float* dinv_gw = ws;                 // [N]
    float* dinv_g1 = ws + N;             // [N]
    float* dinv_iw = ws + 2 * N;         // [N]
    float* dinv_i1 = ws + 3 * N;         // [N]
    float* norm_gw = ws + 4 * N;         // [E]
    float* norm_g1 = norm_gw + E;        // [E]
    float* norm_iw = norm_g1 + E;        // [E]
    float* norm_i1 = norm_iw + E;        // [E]
    float* xw      = norm_i1 + E;        // [N*3000]
    float* buf1    = xw + (long)N * 3000;   // [N*512]
    float* buf2    = buf1 + (long)N * 512;  // [N*128]

    // output slots: (h2, i2, c2, h4, i4, c4)
    float* out   = (float*)d_out;
    float* o_h2  = out;
    float* o_i2  = out + (long)N * 64;
    float* o_c2  = out + (long)N * 128;
    float* o_h4  = out + (long)N * 192;
    float* o_i4  = o_h4 + (long)N * 3000;
    float* o_c4  = o_i4 + (long)N * 3000;

    // ---- degrees & edge norms (computed once, reused across layers) ----
    k_fill1<<<ceil_div(4 * N, 256), 256, 0, stream>>>(ws, 4 * N);
    k_deg_accum<<<ceil_div(E, 256), 256, 0, stream>>>(dst_g, gene_attr, dinv_gw, E, 1);
    k_deg_accum<<<ceil_div(E, 256), 256, 0, stream>>>(dst_g, gene_attr, dinv_g1, E, 0);
    k_deg_accum<<<ceil_div(E, 256), 256, 0, stream>>>(dst_i, img_attr,  dinv_iw, E, 1);
    k_deg_accum<<<ceil_div(E, 256), 256, 0, stream>>>(dst_i, img_attr,  dinv_i1, E, 0);
    k_deg_to_dinv<<<ceil_div(4 * N, 256), 256, 0, stream>>>(ws, 4 * N);
    k_edge_norm<<<ceil_div(E, 256), 256, 0, stream>>>(src_g, dst_g, gene_attr, dinv_gw, norm_gw, E, 1);
    k_edge_norm<<<ceil_div(E, 256), 256, 0, stream>>>(src_g, dst_g, gene_attr, dinv_g1, norm_g1, E, 0);
    k_edge_norm<<<ceil_div(E, 256), 256, 0, stream>>>(src_i, dst_i, img_attr,  dinv_iw, norm_iw, E, 1);
    k_edge_norm<<<ceil_div(E, 256), 256, 0, stream>>>(src_i, dst_i, img_attr,  dinv_i1, norm_i1, E, 0);

    // ---- one GCN layer: GEMM -> self-loop+bias -> edge scatter -> (ELU) ----
    auto run_gcn = [&](const float* x, int layer, const int* src, const int* dst,
                       const float* norm, const float* dinv, float* dst_buf, bool elu) {
        int fi = FI[layer], fo = FO[layer];
        dim3 g(ceil_div(N, TM), ceil_div(fo, TN));
        gemm_bf16_wmma<<<g, 128, 0, stream>>>(x, Wp(layer), xw, N, fi, fo);
        long total = (long)N * fo;
        k_init_out<<<(int)((total + 255) / 256), 256, 0, stream>>>(xw, dinv, Bp(layer), dst_buf, N, fo);
        k_scatter<<<ceil_div(E, 8), 256, 0, stream>>>(xw, src, dst, norm, dst_buf, E, fo);
        if (elu) k_elu<<<(int)((total + 255) / 256), 256, 0, stream>>>(dst_buf, total);
    };

    // gene branch
    run_gcn(features, L_conv1, src_g, dst_g, norm_gw, dinv_gw, buf1, true);   // h1
    run_gcn(buf1,     L_conv2, src_g, dst_g, norm_gw, dinv_gw, o_h2, false);  // h2
    run_gcn(o_h2,     L_conv3, src_g, dst_g, norm_g1, dinv_g1, buf1, true);   // h3
    run_gcn(buf1,     L_conv4, src_g, dst_g, norm_g1, dinv_g1, o_h4, false);  // h4
    // image branch
    run_gcn(img_feat, L_img1, src_i, dst_i, norm_iw, dinv_iw, buf1, true);    // i1
    run_gcn(buf1,     L_img2, src_i, dst_i, norm_iw, dinv_iw, o_i2, false);   // i2
    run_gcn(o_i2,     L_img3, src_i, dst_i, norm_i1, dinv_i1, buf1, true);    // i3
    run_gcn(buf1,     L_img4, src_i, dst_i, norm_i1, dinv_i1, o_i4, false);   // i4
    // neck
    k_concat<<<ceil_div(N * 128, 256), 256, 0, stream>>>(o_h2, o_i2, buf2, N);
    run_gcn(buf2, L_neck,  src_g, dst_g, norm_g1, dinv_g1, buf1, true);       // combine
    run_gcn(buf1, L_neck2, src_g, dst_g, norm_g1, dinv_g1, o_c2, false);      // c2
    run_gcn(o_c2, L_c3,    src_g, dst_g, norm_g1, dinv_g1, buf1, true);       // c3
    run_gcn(buf1, L_c4,    src_g, dst_g, norm_g1, dinv_g1, o_c4, false);      // c4

    (void)in_sizes; (void)n_in; (void)out_size; (void)ws_size;
}
